// MultiHeadAttention_20151986553479
// MI455X (gfx1250) — matrix operand order repaired and checked by reading, not compiled, not run on hardware
//
#include <hip/hip_runtime.h>
#include <hip/hip_bf16.h>

// MI455X / gfx1250 fused QKV + multi-head attention, bf16 WMMA with f32 accumulate.
// B=2, S=2048, D=1024, H=16, AD=64.

typedef __attribute__((ext_vector_type(16))) __bf16        v16bf;
typedef __attribute__((ext_vector_type(8)))  float         v8f;
typedef __attribute__((ext_vector_type(4)))  unsigned int  u32x4;
typedef __attribute__((ext_vector_type(4)))  int           v4i;

#define B_   2
#define S_   2048
#define D_   1024
#define H_   16
#define AD_  64
#define N3D_ 3072  // 3*D

#if defined(__has_builtin)
#if __has_builtin(__builtin_amdgcn_global_load_async_to_lds_b128)
#define HAVE_ASYNC_LDS 1
#endif
#endif

static __device__ __forceinline__ unsigned short f2bf(float f) {
    unsigned int u = __float_as_uint(f);
    u += 0x7FFFu + ((u >> 16) & 1u);   // round-to-nearest-even
    return (unsigned short)(u >> 16);
}

union FragAB { u32x4 u[2]; v16bf v; };   // 32 bytes/lane: one 16x32 bf16 A or 32x16 bf16 B fragment
union Acc    { v8f v; float f[8]; };     // 16x16 f32 C/D fragment

static __device__ __forceinline__ v8f wmma_bf16(v16bf a, v16bf b, v8f c) {
    // D = A(16x32 bf16) * B(32x16 bf16) + C(16x16 f32)
    return __builtin_amdgcn_wmma_f32_16x16x32_bf16(false, a, false, b, (short)0, c, false, false);
}

static __device__ __forceinline__ u32x4 ld16(const unsigned short* p) {
    return *(const u32x4*)p;
}

#ifdef HAVE_ASYNC_LDS
// Builtin signature (from hipcc diagnostic): (v4i AS1*, v4i AS3*, imm int, imm int)
typedef __attribute__((address_space(1))) v4i g_v4i;   // global int4
typedef __attribute__((address_space(3))) v4i l_v4i;   // LDS int4

// Async DMA: global (AS1) -> LDS (AS3), 16 bytes per lane, tracked by ASYNCcnt.
static __device__ __forceinline__ void async_b128(const unsigned short* g, unsigned short* l) {
    void* gv = (void*)g;   // drop const
    void* lv = (void*)l;
    __builtin_amdgcn_global_load_async_to_lds_b128((g_v4i*)gv, (l_v4i*)lv, 0, 0);
}
static __device__ __forceinline__ void wait_async0() {
#if __has_builtin(__builtin_amdgcn_s_wait_asynccnt)
    __builtin_amdgcn_s_wait_asynccnt(0);
#else
    asm volatile("s_wait_asynccnt 0x0" ::: "memory");
#endif
}
#endif

// ---------------------------------------------------------------- converts --
__global__ void cvt_x_kernel(const float* __restrict__ x, unsigned short* __restrict__ xbf, int n) {
    int i = blockIdx.x * blockDim.x + threadIdx.x;
    if (i < n) xbf[i] = f2bf(x[i]);
}

// W[k][n] (D x 3D) -> Wt[n][k] (3D x D) bf16, so GEMM B-columns are K-contiguous.
__global__ void cvt_wt_kernel(const float* __restrict__ W, unsigned short* __restrict__ wt) {
    int i = blockIdx.x * blockDim.x + threadIdx.x;
    if (i < N3D_ * D_) {
        int n = i >> 10;        // / D_
        int k = i & (D_ - 1);
        wt[i] = f2bf(W[(size_t)k * N3D_ + n]);
    }
}

// --------------------------------------------------------------- QKV GEMM ---
// C[4096 x 3072] = Xbf[4096 x 1024] @ Wt^T + bias, scattered to Q/K/Vt bf16.
// Block: 256 threads = 8 waves as 2(M) x 4(N); wave tile 32x32 (4 accumulators).
__global__ __launch_bounds__(256) void qkv_gemm_kernel(
    const unsigned short* __restrict__ Xbf,
    const unsigned short* __restrict__ Wt,
    const float* __restrict__ bias,
    unsigned short* __restrict__ Qbf,
    unsigned short* __restrict__ Kbf,
    unsigned short* __restrict__ Vtbf)
{
    const int lane   = threadIdx.x & 31;
    const int wave   = threadIdx.x >> 5;
    const int l15    = lane & 15;
    const int half8  = (lane >> 4) << 3;   // 0 | 8
    const int half16 = (lane >> 4) << 4;   // 0 | 16
    const int m0 = blockIdx.y * 64  + (wave >> 2) * 32;
    const int n0 = blockIdx.x * 128 + (wave & 3) * 32;

    Acc acc[2][2];
#pragma unroll
    for (int i = 0; i < 2; ++i)
#pragma unroll
        for (int j = 0; j < 2; ++j)
#pragma unroll
            for (int r = 0; r < 8; ++r) acc[i][j].f[r] = 0.0f;

    for (int kk = 0; kk < D_; kk += 32) {
        FragAB a[2], b[2];
#pragma unroll
        for (int i = 0; i < 2; ++i) {
            // A fragment: row = m, chunks at K = kk+half8 and kk+16+half8
            const unsigned short* ap = Xbf + (size_t)(m0 + i * 16 + l15) * D_ + kk + half8;
            a[i].u[0] = ld16(ap);
            a[i].u[1] = ld16(ap + 16);
        }
#pragma unroll
        for (int j = 0; j < 2; ++j) {
            // B fragment: col = n, chunks at K = kk+half8 and kk+16+half8
            const unsigned short* bp = Wt + (size_t)(n0 + j * 16 + l15) * D_ + kk + half8;
            b[j].u[0] = ld16(bp);
            b[j].u[1] = ld16(bp + 16);
        }
#pragma unroll
        for (int i = 0; i < 2; ++i)
#pragma unroll
            for (int j = 0; j < 2; ++j)
                acc[i][j].v = wmma_bf16(a[i].v, b[j].v, acc[i][j].v);
    }

    // Epilogue: bias add, bf16 convert, scatter to Q/K/Vt.
#pragma unroll
    for (int j = 0; j < 2; ++j) {
        const int n  = n0 + j * 16 + l15;
        const float bn = bias[n];
        const int h  = n / (3 * AD_);       // 192 per head
        const int jj = n % (3 * AD_);       // 16-col tile never crosses q/k/v boundary
#pragma unroll
        for (int i = 0; i < 2; ++i) {
#pragma unroll
            for (int r = 0; r < 8; ++r) {
                const int m = m0 + i * 16 + r + half8;
                const int bb = m >> 11;           // / S_
                const int s  = m & (S_ - 1);
                const size_t bh = (size_t)bb * H_ + h;
                const unsigned short bv = f2bf(acc[i][j].f[r] + bn);
                if (jj < AD_)            Qbf [(bh * S_ + s) * AD_ + jj]               = bv;
                else if (jj < 2 * AD_)   Kbf [(bh * S_ + s) * AD_ + (jj - AD_)]       = bv;
                else                     Vtbf[(bh * AD_ + (jj - 2 * AD_)) * S_ + s]   = bv;  // V transposed
            }
        }
    }
}

// ---------------------------------------------------------- flash attention --
// grid = (S/128, B*H); block = 256 threads = 8 waves; each wave: 16 query rows.
// K/V tiles for the 32-key block are staged into LDS once per block via async
// global->LDS DMA (ASYNCcnt) and shared by all 8 waves.
__global__ __launch_bounds__(256) void flash_attn_kernel(
    const unsigned short* __restrict__ Qbf,
    const unsigned short* __restrict__ Kbf,
    const unsigned short* __restrict__ Vtbf,
    float* __restrict__ out)
{
    __shared__ __align__(16) unsigned short ktile[32 * 64];    // [key][d]   4 KB
    __shared__ __align__(16) unsigned short vtile[64 * 32];    // [d][key]   4 KB
    __shared__ __align__(16) unsigned short pbuf[8][16 * 32];  // per-wave P bounce, 8 KB

    const int tid    = threadIdx.x;
    const int lane   = tid & 31;
    const int wave   = tid >> 5;
    const int l15    = lane & 15;
    const int half8  = (lane >> 4) << 3;
    const int half16 = (lane >> 4) << 4;
    const int bh = blockIdx.y;
    const int q0 = blockIdx.x * 128 + wave * 16;

    const unsigned short* qb  = Qbf  + (size_t)bh * S_ * AD_;
    const unsigned short* kg  = Kbf  + (size_t)bh * S_ * AD_;
    const unsigned short* vg  = Vtbf + (size_t)bh * AD_ * S_;

    // Q fragments (16 x 64 = two K-steps), loaded once from global.
    FragAB qf[2];
#pragma unroll
    for (int t = 0; t < 2; ++t) {
        const unsigned short* p = qb + (size_t)(q0 + l15) * AD_ + t * 32 + half8;
        qf[t].u[0] = ld16(p);
        qf[t].u[1] = ld16(p + 16);
    }

    Acc o[4];
#pragma unroll
    for (int t = 0; t < 4; ++t)
#pragma unroll
        for (int r = 0; r < 8; ++r) o[t].f[r] = 0.0f;

    float mrun[8], lrun[8];
#pragma unroll
    for (int r = 0; r < 8; ++r) { mrun[r] = -3.0e38f; lrun[r] = 0.0f; }

    const float scale = 0.125f;  // 1/sqrt(64)

    for (int kb = 0; kb < S_; kb += 32) {
        __syncthreads();   // all readers of the previous tile are done

        // ---- cooperative stage: K tile (contiguous 4 KB) + Vt tile (64 rows x 64 B) ----
        {
            const unsigned short* gk = kg + (size_t)kb * AD_ + tid * 8;       // chunk tid
            const int dd = tid >> 2, ch = tid & 3;
            const unsigned short* gv = vg + (size_t)dd * S_ + kb + ch * 8;
#ifdef HAVE_ASYNC_LDS
            async_b128(gk, &ktile[tid * 8]);
            async_b128(gv, &vtile[dd * 32 + ch * 8]);
            wait_async0();
#else
            *(u32x4*)&ktile[tid * 8]           = ld16(gk);
            *(u32x4*)&vtile[dd * 32 + ch * 8]  = ld16(gv);
#endif
        }
        __syncthreads();

        // ---- scores S = Q @ K^T : two 16x16 tiles over 32 keys (K frags from LDS) ----
        Acc sc[2];
#pragma unroll
        for (int t2 = 0; t2 < 2; ++t2) {
            FragAB kf0, kf1;
            const unsigned short* kp = ktile + (t2 * 16 + l15) * AD_;
            kf0.u[0] = ld16(kp + half8);
            kf0.u[1] = ld16(kp + half8 + 16);
            kf1.u[0] = ld16(kp + 32 + half8);
            kf1.u[1] = ld16(kp + 32 + half8 + 16);
            Acc z;
#pragma unroll
            for (int r = 0; r < 8; ++r) z.f[r] = 0.0f;
            v8f c = z.v;
            c = wmma_bf16(qf[0].v, kf0.v, c);
            c = wmma_bf16(qf[1].v, kf1.v, c);
            sc[t2].v = c;
        }

        // ---- online softmax (rows live in 16-lane halves) ----
        float rescale[8];
#pragma unroll
        for (int r = 0; r < 8; ++r) {
            float a  = sc[0].f[r] * scale;
            float b2 = sc[1].f[r] * scale;
            float mx = fmaxf(a, b2);
#pragma unroll
            for (int off = 8; off >= 1; off >>= 1) mx = fmaxf(mx, __shfl_xor(mx, off, 32));
            const float mnew = fmaxf(mrun[r], mx);
            const float rs = __expf(mrun[r] - mnew);
            const float p0 = __expf(a  - mnew);
            const float p1 = __expf(b2 - mnew);
            sc[0].f[r] = p0;
            sc[1].f[r] = p1;
            float su = p0 + p1;
#pragma unroll
            for (int off = 8; off >= 1; off >>= 1) su += __shfl_xor(su, off, 32);
            lrun[r] = lrun[r] * rs + su;
            mrun[r] = mnew;
            rescale[r] = rs;
        }
#pragma unroll
        for (int t = 0; t < 4; ++t)
#pragma unroll
            for (int r = 0; r < 8; ++r) o[t].f[r] *= rescale[r];

        // ---- P (16x32 bf16): bounce D-layout -> A-layout through per-wave LDS ----
        unsigned short* pb = pbuf[wave];
#pragma unroll
        for (int t2 = 0; t2 < 2; ++t2)
#pragma unroll
            for (int r = 0; r < 8; ++r)
                pb[(r + half8) * 32 + t2 * 16 + l15] = f2bf(sc[t2].f[r]);
        asm volatile("s_wait_dscnt 0" ::: "memory");

        FragAB pf;
        const unsigned short* pp = pb + l15 * 32 + half8;
        pf.u[0] = ld16(pp);
        pf.u[1] = ld16(pp + 16);

        // ---- O += P @ V : V^T rows are K-contiguous (V frags from LDS) ----
#pragma unroll
        for (int t = 0; t < 4; ++t) {
            FragAB vf;
            const unsigned short* vp = vtile + (t * 16 + l15) * 32 + half8;
            vf.u[0] = ld16(vp);
            vf.u[1] = ld16(vp + 16);
            o[t].v = wmma_bf16(pf.v, vf.v, o[t].v);
        }
    }

    // ---- normalize + store (output flat layout == [B,H,S,AD]) ----
#pragma unroll
    for (int r = 0; r < 8; ++r) {
        const float inv = 1.0f / lrun[r];
        const int row = q0 + r + half8;
        float* op = out + ((size_t)bh * S_ + row) * AD_;
#pragma unroll
        for (int t = 0; t < 4; ++t)
            op[t * 16 + l15] = o[t].f[r] * inv;
    }
}

// ------------------------------------------------------------------ launch --
extern "C" void kernel_launch(void* const* d_in, const int* in_sizes, int n_in,
                              void* d_out, int out_size, void* d_ws, size_t ws_size,
                              hipStream_t stream) {
    const float* x    = (const float*)d_in[0];   // [B,S,D]
    const float* Wqkv = (const float*)d_in[1];   // [D,3D]
    const float* bqkv = (const float*)d_in[2];   // [3D]
    float* out = (float*)d_out;

    // Workspace carve-up (all bf16 as ushort): 38 MB total.
    unsigned short* Xbf  = (unsigned short*)d_ws;               // 4096*1024
    unsigned short* Wt   = Xbf  + (size_t)(B_ * S_) * D_;       // 3072*1024
    unsigned short* Qbf  = Wt   + (size_t)N3D_ * D_;            // B*H*S*AD
    unsigned short* Kbf  = Qbf  + (size_t)B_ * H_ * S_ * AD_;
    unsigned short* Vtbf = Kbf  + (size_t)B_ * H_ * S_ * AD_;

    const int nx = B_ * S_ * D_;
    cvt_x_kernel<<<(nx + 255) / 256, 256, 0, stream>>>(x, Xbf, nx);

    const int nw = N3D_ * D_;
    cvt_wt_kernel<<<(nw + 255) / 256, 256, 0, stream>>>(Wqkv, Wt);

    dim3 ggrid(N3D_ / 128, (B_ * S_) / 64);   // (24, 64)
    qkv_gemm_kernel<<<ggrid, 256, 0, stream>>>(Xbf, Wt, bqkv, Qbf, Kbf, Vtbf);

    dim3 agrid(S_ / 128, B_ * H_);            // (16, 32)
    flash_attn_kernel<<<agrid, 256, 0, stream>>>(Qbf, Kbf, Vtbf, out);
}
